// Local2FWLGAT_12051678233186
// MI455X (gfx1250) — compile-verified, matrix operand order checked by reading, not run on hardware
//
#include <hip/hip_runtime.h>
#include <hip/hip_bf16.h>

// ---------------------------------------------------------------------------
// CDNA5 / gfx1250 implementation of the Local2FWL + GAT network.
// All dense MLPs run on v_wmma_f32_16x16x32_f16 (wave32 WMMA).
// 16 rows per 256-thread block; 8 waves each own a 16-column slab of the
// 128-wide output. A-fragments come from LDS (two b128 ds loads per lane),
// B-fragments are pre-packed per-lane in scratch (one 32B load per lane).
// f16 activation mirrors (hn16/hp16) halve gather bytes and make staging
// pure 16B global->LDS copies. Fast-math rcp in all epilogues.
// ---------------------------------------------------------------------------

typedef __attribute__((ext_vector_type(16))) _Float16 v16h;
typedef __attribute__((ext_vector_type(8)))  _Float16 v8h;
typedef __attribute__((ext_vector_type(4)))  _Float16 v4h;
typedef __attribute__((ext_vector_type(8)))  float    v8f;
typedef __attribute__((ext_vector_type(4)))  float    v4f;

#define HID   128
#define HEADS 4
#define HDIM  32
#define NEG_SLOPE 0.2f

// -------------------------- WMMA core helpers ------------------------------

// A-fragment for V_WMMA_F32_16X16X32_F16 from a row-major f16 LDS tile.
__device__ __forceinline__ v16h frag_from_rowmajor(const _Float16* A, int ldk,
                                                   int lane, int kt) {
  int row = lane & 15;
  const _Float16* p = A + row * ldk + kt * 32 + ((lane >> 4) << 3);
  v8h lo = *(const v8h*)p;
  v8h hi = *(const v8h*)(p + 16);
  v16h r;
#pragma unroll
  for (int i = 0; i < 8; ++i) { r[i] = lo[i]; r[8 + i] = hi[i]; }
  return r;
}

// One 16x16 output tile accumulated over `ktiles` K-tiles of 32.
__device__ __forceinline__ v8f wmma_layer(const _Float16* A, int ldk, int ktiles,
                                          const v16h* __restrict__ Wf,
                                          int slab, int lane) {
  v8f c = {};
  for (int kt = 0; kt < ktiles; ++kt) {
    v16h a = frag_from_rowmajor(A, ldk, lane, kt);
    v16h b = Wf[(size_t)(kt * 8 + slab) * 32 + lane];
    if (kt + 1 < ktiles)  // hide next B-fragment latency (global_prefetch_b8)
      __builtin_prefetch(&Wf[(size_t)((kt + 1) * 8 + slab) * 32 + lane], 0, 3);
    c = __builtin_amdgcn_wmma_f32_16x16x32_f16(false, a, false, b,
                                               (short)0, c, false, false);
  }
  return c;
}

// Fast sigmoid-based SiLU: one v_exp_f32 + v_rcp_f32 per element.
__device__ __forceinline__ float fast_silu(float v) {
  return v * __builtin_amdgcn_rcpf(1.0f + __expf(-v));
}

// Apply bias + SiLU to a C fragment and write the hidden 16x128 f16 tile to LDS.
// C layout: VGPR r, lane l -> row = r + 8*(l>>4), col = slab*16 + (l&15).
__device__ __forceinline__ void silu_bias_to_lds(v8f c, const float* __restrict__ b1,
                                                 _Float16* Hs, int slab, int lane) {
  int col = slab * 16 + (lane & 15);
  float bias = b1[col];
  int rbase = (lane >> 4) << 3;
#pragma unroll
  for (int r = 0; r < 8; ++r) {
    float s = fast_silu(c[r] + bias);
    Hs[(rbase + r) * HID + col] = (_Float16)s;
  }
}

__device__ __forceinline__ void atomicMaxF(float* addr, float val) {
  int* ai = (int*)addr;
  int cur = *ai;
  while (__int_as_float(cur) < val) {
    int prev = atomicCAS(ai, cur, __float_as_int(val));
    if (prev == cur) break;
    cur = prev;
  }
}

// -------------------------- weight fragment packing ------------------------
__global__ void __launch_bounds__(256)
l2f_pack(const float* __restrict__ W, int Kreal, int ktiles,
         _Float16* __restrict__ dst) {
  int tid = blockIdx.x * 256 + threadIdx.x;
  int total = ktiles * 256;            // ktiles * 8 slabs * 32 lanes
  if (tid >= total) return;
  int lane = tid & 31;
  int slab = (tid >> 5) & 7;
  int kt   = tid >> 8;
  int col  = slab * 16 + (lane & 15);
  int kbase = kt * 32 + ((lane >> 4) << 3);
  _Float16* out = dst + (size_t)tid * 16;
#pragma unroll
  for (int h = 0; h < 16; ++h) {
    int k = kbase + ((h >> 3) << 4) + (h & 7);
    float v = (k < Kreal) ? W[(size_t)k * HID + col] : 0.0f;
    out[h] = (_Float16)v;
  }
}

// -------------------------- generic fill -----------------------------------
__global__ void l2f_fill(float* __restrict__ p, long long n, float v) {
  long long i = (long long)blockIdx.x * 256 + threadIdx.x;
  long long stride = (long long)gridDim.x * 256;
  for (; i < n; i += stride) p[i] = v;
}

// -------------------------- node encoder: 16 -> 128 -> 128 -----------------
__global__ void __launch_bounds__(256)
l2f_node_enc(const float* __restrict__ x, int N,
             const v16h* __restrict__ W1, const float* __restrict__ b1,
             const v16h* __restrict__ W2, const float* __restrict__ b2,
             float* __restrict__ h_node, _Float16* __restrict__ hn16) {
  __shared__ __align__(16) _Float16 sA[16 * 32];
  __shared__ __align__(16) _Float16 sH[16 * HID];
  int blk = blockIdx.x;
  for (int e = threadIdx.x; e < 16 * 32; e += 256) {
    int row = e >> 5, k = e & 31;
    int g = blk * 16 + row;
    float v = (k < 16 && g < N) ? x[(size_t)g * 16 + k] : 0.0f;
    sA[e] = (_Float16)v;
  }
  __syncthreads();
  int lane = threadIdx.x & 31, slab = threadIdx.x >> 5;
  v8f c1 = wmma_layer(sA, 32, 1, W1, slab, lane);
  silu_bias_to_lds(c1, b1, sH, slab, lane);
  __syncthreads();
  v8f c2 = wmma_layer(sH, HID, 4, W2, slab, lane);
  int col = slab * 16 + (lane & 15);
  int rbase = (lane >> 4) << 3;
  float bb = b2[col];
#pragma unroll
  for (int r = 0; r < 8; ++r) {
    int g = blk * 16 + rbase + r;
    if (g < N) {
      float o = c2[r] + bb;
      h_node[(size_t)g * HID + col] = o;
      hn16[(size_t)g * HID + col] = (_Float16)o;
    }
  }
}

// -------------------------- pair init: 258 -> 128 -> 128 -------------------
__global__ void __launch_bounds__(256)
l2f_pair_init(const int* __restrict__ pidx, const float* __restrict__ dist,
              const float* __restrict__ bflag, const _Float16* __restrict__ hn16,
              int P,
              const v16h* __restrict__ W1, const float* __restrict__ b1,
              const v16h* __restrict__ W2, const float* __restrict__ b2,
              float* __restrict__ h_pair, _Float16* __restrict__ hp16) {
  __shared__ __align__(16) _Float16 sA[16 * 288];
  __shared__ __align__(16) _Float16 sH[16 * HID];
  int blk = blockIdx.x;
  // k < 256: 16B chunks from the f16 node mirror (gather of hn16[a] | hn16[b])
  for (int c = threadIdx.x; c < 16 * 32; c += 256) {
    int row = c >> 5, cj = c & 31;
    int p = blk * 16 + row;
    v8h val = {};
    if (p < P) {
      int src = (cj < 16) ? pidx[p] : pidx[P + p];
      int off = (cj & 15) << 3;
      val = *(const v8h*)(&hn16[(size_t)src * HID + off]);
    }
    *(v8h*)(&sA[row * 288 + (cj << 3)]) = val;
  }
  // tail k in [256,288): dist, flag, zero pad
  for (int e = threadIdx.x; e < 16 * 32; e += 256) {
    int row = e >> 5, k = (e & 31) + 256;
    int p = blk * 16 + row;
    float v = 0.0f;
    if (p < P) {
      if (k == 256) v = dist[p];
      else if (k == 257) v = bflag[p];
    }
    sA[row * 288 + k] = (_Float16)v;
  }
  __syncthreads();
  int lane = threadIdx.x & 31, slab = threadIdx.x >> 5;
  v8f c1 = wmma_layer(sA, 288, 9, W1, slab, lane);
  silu_bias_to_lds(c1, b1, sH, slab, lane);
  __syncthreads();
  v8f c2 = wmma_layer(sH, HID, 4, W2, slab, lane);
  int col = slab * 16 + (lane & 15);
  int rbase = (lane >> 4) << 3;
  float bb = b2[col];
#pragma unroll
  for (int r = 0; r < 8; ++r) {
    int p = blk * 16 + rbase + r;
    if (p < P) {
      float o = c2[r] + bb;
      h_pair[(size_t)p * HID + col] = o;
      hp16[(size_t)p * HID + col] = (_Float16)o;
    }
  }
}

// -------------------------- FWL psi: 388 -> 128 -> 128, scatter-add --------
__global__ void __launch_bounds__(256)
l2f_psi(const int* __restrict__ vu, const int* __restrict__ uw,
        const int* __restrict__ vw, const float* __restrict__ geom,
        const _Float16* __restrict__ hp16, int T,
        const v16h* __restrict__ W1, const float* __restrict__ b1,
        const v16h* __restrict__ W2, const float* __restrict__ b2,
        float* __restrict__ agg) {
  __shared__ __align__(16) _Float16 sA[16 * 416];
  __shared__ __align__(16) _Float16 sH[16 * HID];
  int blk = blockIdx.x;
  // k < 384: 16B chunks from the f16 pair mirror (hp16[vu]|hp16[uw]|hp16[vw])
  for (int c = threadIdx.x; c < 16 * 48; c += 256) {
    int row = c / 48, cj = c - row * 48;
    int t = blk * 16 + row;
    v8h val = {};
    if (t < T) {
      int which = cj >> 4;                 // 0,1,2
      int off = (cj & 15) << 3;            // 0..120
      int src = (which == 0) ? vu[t] : (which == 1) ? uw[t] : vw[t];
      val = *(const v8h*)(&hp16[(size_t)src * HID + off]);
    }
    *(v8h*)(&sA[row * 416 + (cj << 3)]) = val;
  }
  // tail k in [384,416): geom + zero pad
  for (int e = threadIdx.x; e < 16 * 32; e += 256) {
    int row = e >> 5, k = (e & 31) + 384;
    int t = blk * 16 + row;
    float v = 0.0f;
    if (t < T && k < 388) v = geom[(size_t)t * 4 + (k - 384)];
    sA[row * 416 + k] = (_Float16)v;
  }
  __syncthreads();
  int lane = threadIdx.x & 31, slab = threadIdx.x >> 5;
  v8f c1 = wmma_layer(sA, 416, 13, W1, slab, lane);
  silu_bias_to_lds(c1, b1, sH, slab, lane);
  __syncthreads();
  v8f c2 = wmma_layer(sH, HID, 4, W2, slab, lane);
  int col = slab * 16 + (lane & 15);
  int rbase = (lane >> 4) << 3;
  float bb = b2[col];
#pragma unroll
  for (int r = 0; r < 8; ++r) {
    int t = blk * 16 + rbase + r;
    if (t < T) atomicAdd(&agg[(size_t)vw[t] * HID + col], c2[r] + bb);
  }
}

// ---------- FWL phi: 256 -> 128 -> 128, residual + node scatter ------------
__global__ void __launch_bounds__(256)
l2f_phi(const int* __restrict__ pidx, const float* __restrict__ agg,
        float* __restrict__ hp, _Float16* __restrict__ hp16, int P,
        const v16h* __restrict__ W1, const float* __restrict__ b1,
        const v16h* __restrict__ W2, const float* __restrict__ b2,
        float* __restrict__ nmsg) {
  __shared__ __align__(16) _Float16 sA[16 * 256];
  __shared__ __align__(16) _Float16 sH[16 * HID];
  int blk = blockIdx.x;
  // k < 128: straight 16B copies from the f16 pair mirror
  for (int c = threadIdx.x; c < 16 * 16; c += 256) {
    int row = c >> 4, off = (c & 15) << 3;
    int p = blk * 16 + row;
    v8h val = {};
    if (p < P) val = *(const v8h*)(&hp16[(size_t)p * HID + off]);
    *(v8h*)(&sA[row * 256 + off]) = val;
  }
  // k in [128,256): float4 loads of agg, packed b64 LDS stores
  for (int c = threadIdx.x; c < 16 * 32; c += 256) {
    int row = c >> 5, off = (c & 31) << 2;
    int p = blk * 16 + row;
    v4f f = {};
    if (p < P) f = *(const v4f*)(&agg[(size_t)p * HID + off]);
    v4h hv = { (_Float16)f[0], (_Float16)f[1], (_Float16)f[2], (_Float16)f[3] };
    *(v4h*)(&sA[row * 256 + 128 + off]) = hv;
  }
  __syncthreads();
  int lane = threadIdx.x & 31, slab = threadIdx.x >> 5;
  v8f c1 = wmma_layer(sA, 256, 8, W1, slab, lane);
  silu_bias_to_lds(c1, b1, sH, slab, lane);
  __syncthreads();
  v8f c2 = wmma_layer(sH, HID, 4, W2, slab, lane);
  int col = slab * 16 + (lane & 15);
  int rbase = (lane >> 4) << 3;
  float bb = b2[col];
#pragma unroll
  for (int r = 0; r < 8; ++r) {
    int p = blk * 16 + rbase + r;
    if (p < P) {
      float v = hp[(size_t)p * HID + col] + c2[r] + bb;
      hp[(size_t)p * HID + col] = v;
      hp16[(size_t)p * HID + col] = (_Float16)v;
      atomicAdd(&nmsg[(size_t)pidx[p] * HID + col], v);
      atomicAdd(&nmsg[(size_t)pidx[P + p] * HID + col], v);
    }
  }
}

// ---------------- node fuse: 256 -> 128 -> 128, residual -------------------
__global__ void __launch_bounds__(256)
l2f_node_fuse(const float* __restrict__ nmsg, float* __restrict__ hn,
              _Float16* __restrict__ hn16, int N,
              const v16h* __restrict__ W1, const float* __restrict__ b1,
              const v16h* __restrict__ W2, const float* __restrict__ b2) {
  __shared__ __align__(16) _Float16 sA[16 * 256];
  __shared__ __align__(16) _Float16 sH[16 * HID];
  int blk = blockIdx.x;
  for (int c = threadIdx.x; c < 16 * 16; c += 256) {
    int row = c >> 4, off = (c & 15) << 3;
    int n = blk * 16 + row;
    v8h val = {};
    if (n < N) val = *(const v8h*)(&hn16[(size_t)n * HID + off]);
    *(v8h*)(&sA[row * 256 + off]) = val;
  }
  for (int c = threadIdx.x; c < 16 * 32; c += 256) {
    int row = c >> 5, off = (c & 31) << 2;
    int n = blk * 16 + row;
    v4f f = {};
    if (n < N) f = *(const v4f*)(&nmsg[(size_t)n * HID + off]);
    v4h hv = { (_Float16)f[0], (_Float16)f[1], (_Float16)f[2], (_Float16)f[3] };
    *(v4h*)(&sA[row * 256 + 128 + off]) = hv;
  }
  __syncthreads();
  int lane = threadIdx.x & 31, slab = threadIdx.x >> 5;
  v8f c1 = wmma_layer(sA, 256, 8, W1, slab, lane);
  silu_bias_to_lds(c1, b1, sH, slab, lane);
  __syncthreads();
  v8f c2 = wmma_layer(sH, HID, 4, W2, slab, lane);
  int col = slab * 16 + (lane & 15);
  int rbase = (lane >> 4) << 3;
  float bb = b2[col];
#pragma unroll
  for (int r = 0; r < 8; ++r) {
    int n = blk * 16 + rbase + r;
    if (n < N) {
      float o = hn[(size_t)n * HID + col] + c2[r] + bb;
      hn[(size_t)n * HID + col] = o;
      hn16[(size_t)n * HID + col] = (_Float16)o;
    }
  }
}

// ---------------- GAT projection: 128x128 GEMM (no act, no bias) -----------
__global__ void __launch_bounds__(256)
l2f_gemm128(const _Float16* __restrict__ in16, int rows,
            const v16h* __restrict__ W, float* __restrict__ out) {
  __shared__ __align__(16) _Float16 sA[16 * 128];
  int blk = blockIdx.x;
  for (int c = threadIdx.x; c < 16 * 16; c += 256) {
    int row = c >> 4, off = (c & 15) << 3;
    int g = blk * 16 + row;
    v8h val = {};
    if (g < rows) val = *(const v8h*)(&in16[(size_t)g * HID + off]);
    *(v8h*)(&sA[row * 128 + off]) = val;
  }
  __syncthreads();
  int lane = threadIdx.x & 31, slab = threadIdx.x >> 5;
  v8f c = wmma_layer(sA, 128, 4, W, slab, lane);
  int col = slab * 16 + (lane & 15);
  int rbase = (lane >> 4) << 3;
#pragma unroll
  for (int r = 0; r < 8; ++r) {
    int g = blk * 16 + rbase + r;
    if (g < rows) out[(size_t)g * HID + col] = c[r];
  }
}

// ---------------- GAT attention coefficients a_s/a_d -----------------------
__global__ void __launch_bounds__(256)
l2f_attn(const float* __restrict__ xh, const float* __restrict__ as_w,
         const float* __restrict__ ad_w, int N,
         float* __restrict__ a_s, float* __restrict__ a_d) {
  int tid = blockIdx.x * 256 + threadIdx.x;
  if (tid >= N * HEADS) return;
  int n = tid >> 2, h = tid & 3;
  const float* row = xh + (size_t)n * HID + h * HDIM;
  float s = 0.f, d = 0.f;
  for (int i = 0; i < HDIM; ++i) {
    s += row[i] * as_w[h * HDIM + i];
    d += row[i] * ad_w[h * HDIM + i];
  }
  a_s[tid] = s;
  a_d[tid] = d;
}

// ---------------- GAT edge passes ------------------------------------------
__global__ void __launch_bounds__(256)
l2f_edge_max(const int* __restrict__ eidx, int E, int N,
             const float* __restrict__ a_s, const float* __restrict__ a_d,
             float* __restrict__ m, float* __restrict__ ebuf) {
  int tid = blockIdx.x * 256 + threadIdx.x;
  int total = (E + N) * HEADS;
  if (tid >= total) return;
  int e = tid >> 2, h = tid & 3;
  int s, d;
  if (e < E) { s = eidx[e]; d = eidx[E + e]; } else { s = d = e - E; }
  float v = a_s[s * HEADS + h] + a_d[d * HEADS + h];
  v = (v > 0.f) ? v : NEG_SLOPE * v;
  ebuf[tid] = v;
  atomicMaxF(&m[d * HEADS + h], v);
}

__global__ void __launch_bounds__(256)
l2f_edge_wsum(const int* __restrict__ eidx, int E, int N,
              const float* __restrict__ m, float* __restrict__ ebuf,
              float* __restrict__ dn) {
  int tid = blockIdx.x * 256 + threadIdx.x;
  int total = (E + N) * HEADS;
  if (tid >= total) return;
  int e = tid >> 2, h = tid & 3;
  int d = (e < E) ? eidx[E + e] : (e - E);
  float w = __expf(ebuf[tid] - m[d * HEADS + h]);
  ebuf[tid] = w;
  atomicAdd(&dn[d * HEADS + h], w);
}

__global__ void __launch_bounds__(256)
l2f_edge_scatter(const int* __restrict__ eidx, int E, int N,
                 const float* __restrict__ ebuf, const float* __restrict__ dn,
                 const float* __restrict__ xh, float* __restrict__ gout) {
  int tid = blockIdx.x * 256 + threadIdx.x;
  int total = (E + N) * HDIM;
  if (tid >= total) return;
  int e = tid >> 5, j = tid & 31;
  int s, d;
  if (e < E) { s = eidx[e]; d = eidx[E + e]; } else { s = d = e - E; }
#pragma unroll
  for (int h = 0; h < HEADS; ++h) {
    float alpha = ebuf[e * HEADS + h] *
                  __builtin_amdgcn_rcpf(dn[d * HEADS + h]);
    atomicAdd(&gout[(size_t)d * HID + h * HDIM + j],
              xh[(size_t)s * HID + h * HDIM + j] * alpha);
  }
}

__global__ void __launch_bounds__(256)
l2f_gat_final(float* __restrict__ hn, _Float16* __restrict__ hn16,
              const float* __restrict__ gout, const float* __restrict__ bias,
              int N) {
  int tid = blockIdx.x * 256 + threadIdx.x;
  if (tid >= N * HID) return;
  float o = hn[tid] + gout[tid] + bias[tid & (HID - 1)];
  hn[tid] = o;
  hn16[tid] = (_Float16)o;   // keep mirror coherent for next GAT projection
}

// ---------------- global mean pool ----------------------------------------
__global__ void __launch_bounds__(256)
l2f_pool(const float* __restrict__ hn, const int* __restrict__ batch, int N,
         float* __restrict__ sums, float* __restrict__ counts) {
  int tid = blockIdx.x * 256 + threadIdx.x;
  if (tid >= N * HID) return;
  int n = tid >> 7, c = tid & 127;
  int g = batch[n];
  atomicAdd(&sums[(size_t)g * HID + c], hn[tid]);
  if (c == 0) atomicAdd(&counts[g], 1.0f);
}

// ---------------- head MLP: 128 -> 128 -> 1 --------------------------------
__global__ void __launch_bounds__(256)
l2f_head(const float* __restrict__ sums, const float* __restrict__ counts, int G,
         const v16h* __restrict__ W1, const float* __restrict__ b1,
         const float* __restrict__ W2v, const float* __restrict__ b2,
         float* __restrict__ out) {
  __shared__ __align__(16) _Float16 sA[16 * 128];
  __shared__ __align__(16) _Float16 sH[16 * HID];
  int blk = blockIdx.x;
  for (int e = threadIdx.x; e < 16 * 128; e += 256) {
    int row = e >> 7, k = e & 127;
    int g = blk * 16 + row;
    float v = 0.0f;
    if (g < G) {
      float cnt = counts[g];
      cnt = (cnt > 1.0f) ? cnt : 1.0f;
      v = sums[(size_t)g * HID + k] * __builtin_amdgcn_rcpf(cnt);
    }
    sA[e] = (_Float16)v;
  }
  __syncthreads();
  int lane = threadIdx.x & 31, slab = threadIdx.x >> 5;
  v8f c1 = wmma_layer(sA, 128, 4, W1, slab, lane);
  silu_bias_to_lds(c1, b1, sH, slab, lane);
  __syncthreads();
  if (threadIdx.x < 16) {
    int g = blk * 16 + threadIdx.x;
    if (g < G) {
      float acc = 0.f;
      for (int d2 = 0; d2 < 128; ++d2)
        acc += (float)sH[threadIdx.x * HID + d2] * W2v[d2];
      out[g] = acc + b2[0];
    }
  }
}

// ---------------------------------------------------------------------------
extern "C" void kernel_launch(void* const* d_in, const int* in_sizes, int n_in,
                              void* d_out, int out_size, void* d_ws, size_t ws_size,
                              hipStream_t stream) {
  const float* x     = (const float*)d_in[0];
  const int*   pidx  = (const int*)  d_in[1];
  const float* dist  = (const float*)d_in[2];
  const float* bflag = (const float*)d_in[3];
  const int*   t_vu  = (const int*)  d_in[4];
  const int*   t_uw  = (const int*)  d_in[5];
  const int*   t_vw  = (const int*)  d_in[6];
  const float* geom  = (const float*)d_in[7];
  const int*   eidx  = (const int*)  d_in[8];
  const int*   batch = (const int*)  d_in[9];
  // d_in[10] = num_graphs (device scalar) -> use out_size instead.

  const int N = in_sizes[0] / 16;
  const int P = in_sizes[2];
  const int T = in_sizes[4];
  const int E = in_sizes[8] / 2;
  const int G = out_size;

  // ---- workspace layout ----
  float* h_pair = (float*)d_ws;                       // P*128 f32
  float* agg    = h_pair + (size_t)P * HID;           // P*128 (FWL), reused in GAT
  float* h_node = agg    + (size_t)P * HID;           // N*128
  float* nmsg   = h_node + (size_t)N * HID;           // N*128 (FWL), reused in GAT/pool
  _Float16* hp16 = (_Float16*)(nmsg + (size_t)N * HID); // P*128 f16 mirror
  _Float16* hn16 = hp16 + (size_t)P * HID;              // N*128 f16 mirror
  _Float16* wp   = hn16 + (size_t)N * HID;              // packed weights

  // GAT-phase aliases (agg/nmsg regions are dead after FWL)
  float* xh    = agg;                                 // N*128
  float* gout  = agg + (size_t)N * HID;               // N*128
  float* a_s   = nmsg;                                // N*4
  float* a_d   = a_s + (size_t)N * HEADS;             // N*4
  float* mbuf  = a_d + (size_t)N * HEADS;             // N*4
  float* dnbuf = mbuf + (size_t)N * HEADS;            // N*4
  float* ebuf  = dnbuf + (size_t)N * HEADS;           // (E+N)*4
  float* sums  = ebuf + (size_t)(E + N) * HEADS;      // G*128
  float* cnts  = sums + (size_t)G * HID;              // G

  // ---- pack all weight matrices into WMMA B-fragment order ----
  size_t woff = 0;
  auto pack = [&](int idx, int Kreal, int ktiles) -> const v16h* {
    _Float16* dst = wp + woff;
    int total = ktiles * 256;
    l2f_pack<<<(total + 255) / 256, 256, 0, stream>>>(
        (const float*)d_in[idx], Kreal, ktiles, dst);
    woff += (size_t)ktiles * 4096;   // ktiles * 256 frags * 16 halves
    return (const v16h*)dst;
  };

  const v16h* encW1  = pack(11, 16, 1);
  const v16h* encW2  = pack(13, 128, 4);
  const v16h* pairW1 = pack(15, 258, 9);
  const v16h* pairW2 = pack(17, 128, 4);
  const v16h *psiW1[3], *psiW2[3], *phiW1[3], *phiW2[3];
  for (int t = 0; t < 3; ++t) {
    psiW1[t] = pack(19 + 8 * t, 388, 13);
    psiW2[t] = pack(21 + 8 * t, 128, 4);
    phiW1[t] = pack(23 + 8 * t, 256, 8);
    phiW2[t] = pack(25 + 8 * t, 128, 4);
  }
  const v16h* fuseW1 = pack(43, 256, 8);
  const v16h* fuseW2 = pack(45, 128, 4);
  const v16h* gatW[2] = { pack(47, 128, 4), pack(51, 128, 4) };
  const v16h* headW1 = pack(55, 128, 4);

  auto bptr = [&](int idx) { return (const float*)d_in[idx]; };

  auto fill = [&](float* p, long long n, float v) {
    long long nb = (n + 255) / 256;
    int grid = (int)((nb < 8192) ? nb : 8192);
    if (grid < 1) grid = 1;
    l2f_fill<<<grid, 256, 0, stream>>>(p, n, v);
  };

  const int nbN = (N + 15) / 16, nbP = (P + 15) / 16;
  const int nbT = (T + 15) / 16, nbG = (G + 15) / 16;

  // ---- node encoder + pair init ----
  l2f_node_enc<<<nbN, 256, 0, stream>>>(x, N, encW1, bptr(12), encW2, bptr(14),
                                        h_node, hn16);
  l2f_pair_init<<<nbP, 256, 0, stream>>>(pidx, dist, bflag, hn16, P,
                                         pairW1, bptr(16), pairW2, bptr(18),
                                         h_pair, hp16);

  // ---- 3 FWL rounds ----
  for (int t = 0; t < 3; ++t) {
    fill(agg, (long long)P * HID, 0.0f);
    fill(nmsg, (long long)N * HID, 0.0f);
    l2f_psi<<<nbT, 256, 0, stream>>>(t_vu, t_uw, t_vw, geom, hp16, T,
                                     psiW1[t], bptr(20 + 8 * t),
                                     psiW2[t], bptr(22 + 8 * t), agg);
    l2f_phi<<<nbP, 256, 0, stream>>>(pidx, agg, h_pair, hp16, P,
                                     phiW1[t], bptr(24 + 8 * t),
                                     phiW2[t], bptr(26 + 8 * t), nmsg);
    l2f_node_fuse<<<nbN, 256, 0, stream>>>(nmsg, h_node, hn16, N,
                                           fuseW1, bptr(44), fuseW2, bptr(46));
  }

  // ---- 2 GAT layers ----
  for (int l = 0; l < 2; ++l) {
    const int base = 47 + 4 * l;
    l2f_gemm128<<<nbN, 256, 0, stream>>>(hn16, N, gatW[l], xh);
    {
      int tot = N * HEADS;
      l2f_attn<<<(tot + 255) / 256, 256, 0, stream>>>(
          xh, bptr(base + 1), bptr(base + 2), N, a_s, a_d);
    }
    fill(mbuf, (long long)N * HEADS, -1e30f);
    fill(dnbuf, (long long)N * HEADS, 0.0f);
    fill(gout, (long long)N * HID, 0.0f);
    {
      int tot = (E + N) * HEADS;
      l2f_edge_max<<<(tot + 255) / 256, 256, 0, stream>>>(eidx, E, N, a_s, a_d,
                                                          mbuf, ebuf);
      l2f_edge_wsum<<<(tot + 255) / 256, 256, 0, stream>>>(eidx, E, N, mbuf,
                                                           ebuf, dnbuf);
    }
    {
      int tot = (E + N) * HDIM;
      l2f_edge_scatter<<<(tot + 255) / 256, 256, 0, stream>>>(
          eidx, E, N, ebuf, dnbuf, xh, gout);
    }
    {
      int tot = N * HID;
      l2f_gat_final<<<(tot + 255) / 256, 256, 0, stream>>>(h_node, hn16, gout,
                                                           bptr(base + 3), N);
    }
  }

  // ---- pool + head ----
  fill(sums, (long long)G * HID, 0.0f);
  fill(cnts, (long long)G, 0.0f);
  {
    int tot = N * HID;
    l2f_pool<<<(tot + 255) / 256, 256, 0, stream>>>(h_node, batch, N, sums, cnts);
  }
  l2f_head<<<nbG, 256, 0, stream>>>(sums, cnts, G, headW1, bptr(56),
                                    bptr(57), bptr(58), (float*)d_out);
}
